// GATNet_24507083391733
// MI455X (gfx1250) — compile-verified, matrix operand order checked by reading, not run on hardware
//
#include <hip/hip_runtime.h>
#include <hip/hip_bf16.h>
#include <math.h>

// ---------------------------------------------------------------- types
typedef __attribute__((ext_vector_type(16))) _Float16 v16h;
typedef __attribute__((ext_vector_type(8)))  _Float16 v8h;
typedef __attribute__((ext_vector_type(8)))  float    v8f;

#define BM 32
#define BN 64
#define BK 64

// problem constants
#define NN   25600
#define BB   128
#define NPG  200
#define EE   128000
#define ETOT (EE + NN)     // edges + self loops
#define FD   78
#define HH   10
#define OD   128
#define EMB  128
#define LL   1000
#define KS   16

// --------------------------------------------- CDNA5 async LDS staging
// GLOBAL_LOAD_ASYNC_TO_LDS_B128: per-lane 16B memory->LDS DMA, ASYNCcnt.
__device__ __forceinline__ void async_ld_b128(unsigned lds_off, const void* g) {
  asm volatile("global_load_async_to_lds_b128 %0, %1, off"
               :: "v"(lds_off), "v"((unsigned long long)(size_t)g)
               : "memory");
}
__device__ __forceinline__ void wait_async0() {
  asm volatile("s_wait_asynccnt 0x0" ::: "memory");
}

// ------------------------------------------------------- WMMA GEMM core
// C[M,N] = act(A[M,K] @ Bt[N,K]^T + epilogue), A/Bt f16, C written f16.
// AMODE 0: A row-major [M,K] (K padded to mult of 64, pads zeroed).
// AMODE 1: implicit conv1d-VALID gemm. row m -> (b,t), col q -> (kpos,ic),
//          A element = in[b][t+kpos][ic], layout [B][Lin][IC], IC % 32 == 0.
// ACT 0: none(+bias)   ACT 1: relu(+bias)   ACT 3: (x+bias)*bn_g*rsq+bn_b, relu
template<int AMODE, int ACT>
__global__ __launch_bounds__(256) void gemm_wmma(
    const _Float16* __restrict__ A, const _Float16* __restrict__ Bt,
    const float* __restrict__ bias, const float* __restrict__ bng,
    const float* __restrict__ bnb, _Float16* __restrict__ C,
    int M, int N, int K, int ldc, int IC, int Lin, int Lout)
{
  __shared__ __align__(16) _Float16 As[BM * BK];   // 4 KB
  __shared__ __align__(16) _Float16 Bs[BN * BK];   // 8 KB

  const int m0 = blockIdx.y * BM;
  const int n0 = blockIdx.x * BN;
  const int t  = threadIdx.x;
  const int wave = t >> 5;
  const int lane = t & 31;
  const int wm  = (wave >> 2) << 4;   // 0 / 16
  const int wn  = (wave & 3) << 4;    // 0,16,32,48
  const int r   = lane & 15;
  const int sel = (lane >> 4) & 1;

  // cooperative tile-load coordinates (16B granules)
  const int arow = t >> 3;            // 0..31
  const int acol = (t & 7) << 3;      // step 8 halfs (16B)
  const int brow = t >> 2;            // 0..63
  const int bcol = (t & 3) << 4;      // step 16 halfs (2x16B)

  // clamp OOB rows to last valid row: garbage feeds only unstored C rows/cols
  int am = m0 + arow; if (am >= M) am = M - 1;
  int bn = n0 + brow; if (bn >= N) bn = N - 1;

  long abase = 0; int conv_bt = 0;
  if (AMODE == 0) {
    abase = (long)am * (long)K;
  } else {
    int b = am / Lout; int tt = am - b * Lout;
    conv_bt = b * Lin + tt;
  }
  const long bbase = (long)bn * (long)K;

  const unsigned ldsA = (unsigned)(size_t)(&As[arow * BK + acol]);
  const unsigned ldsB = (unsigned)(size_t)(&Bs[brow * BK + bcol]);

  v8f acc = {0.f, 0.f, 0.f, 0.f, 0.f, 0.f, 0.f, 0.f};

  for (int k0 = 0; k0 < K; k0 += BK) {
    // ---- async DMA: global -> LDS, no VGPR data, no exec masking ----
    const _Float16* ap;
    if (AMODE == 0) {
      ap = A + abase + k0 + acol;
    } else {
      int q = k0 + acol;
      int kpos = q / IC;
      int ic = q - kpos * IC;
      ap = A + (long)(conv_bt + kpos) * (long)IC + ic;
    }
    async_ld_b128(ldsA, ap);
    const _Float16* bp = Bt + bbase + k0 + bcol;
    async_ld_b128(ldsB, bp);
    async_ld_b128(ldsB + 16, bp + 8);

    wait_async0();
    __syncthreads();

    // ---- two WMMAs per stage (kk = 0, 32) ----
#pragma unroll
    for (int kk = 0; kk < BK; kk += 32) {
      const v8h a_lo = *(const v8h*)(&As[(wm + r) * BK + kk + sel * 8]);
      const v8h a_hi = *(const v8h*)(&As[(wm + r) * BK + kk + sel * 8 + 16]);
      const v8h b_lo = *(const v8h*)(&Bs[(wn + r) * BK + kk + sel * 8]);
      const v8h b_hi = *(const v8h*)(&Bs[(wn + r) * BK + kk + sel * 8 + 16]);
      v16h af = __builtin_shufflevector(a_lo, a_hi, 0,1,2,3,4,5,6,7,8,9,10,11,12,13,14,15);
      v16h bf = __builtin_shufflevector(b_lo, b_hi, 0,1,2,3,4,5,6,7,8,9,10,11,12,13,14,15);
      acc = __builtin_amdgcn_wmma_f32_16x16x32_f16(
          false, af, false, bf, (short)0, acc, false, false);
    }
    __syncthreads();
  }

  // epilogue: VGPR j -> row m0+wm+sel*8+j, col n0+wn+r
  const int col = n0 + wn + r;
  if (col < N) {
    const float RSQ = 0.99999500003749971f;  // 1/sqrt(1+1e-5)
    float bs = 0.f, sc = 0.f, sh = 0.f;
    if (bias) bs = bias[col];
    if (ACT == 3) { sc = bng[col] * RSQ; sh = bnb[col]; }
#pragma unroll
    for (int j = 0; j < 8; ++j) {
      int row = m0 + wm + sel * 8 + j;
      if (row < M) {
        float v = acc[j];
        if (ACT == 0)      { v += bs; }
        else if (ACT == 1) { v += bs; v = v > 0.f ? v : 0.f; }
        else               { v = (v + bs) * sc + sh; v = v > 0.f ? v : 0.f; }
        C[(long)row * ldc + col] = (_Float16)v;
      }
    }
  }
}

// ------------------------------------------------------- helper kernels
__global__ void k_fill_u32(unsigned* p, long n, unsigned v) {
  long i = (long)blockIdx.x * blockDim.x + threadIdx.x;
  if (i < n) p[i] = v;
}
__global__ void k_fill_f32(float* p, long n, float v) {
  long i = (long)blockIdx.x * blockDim.x + threadIdx.x;
  if (i < n) p[i] = v;
}
// f32 [M,K] -> f16 [M,Kp] with zero pad
__global__ void k_cast_pad16(const float* __restrict__ s, _Float16* __restrict__ d,
                             int M, int K, int Kp) {
  long i = (long)blockIdx.x * blockDim.x + threadIdx.x;
  if (i >= (long)M * Kp) return;
  int r = (int)(i / Kp), c = (int)(i - (long)r * Kp);
  d[i] = (c < K) ? (_Float16)s[(long)r * K + c] : (_Float16)0.f;
}
// W f32 [K,N] -> Bt f16 [N,Kp] zero padded
__global__ void k_w_transpose16(const float* __restrict__ w, _Float16* __restrict__ d,
                                int K, int N, int Kp) {
  long i = (long)blockIdx.x * blockDim.x + threadIdx.x;
  if (i >= (long)N * Kp) return;
  int n = (int)(i / Kp), kp = (int)(i - (long)n * Kp);
  d[i] = (kp < K) ? (_Float16)w[(long)kp * N + n] : (_Float16)0.f;
}
// conv weight [OC,IC,KS] -> Bt f16 [OC][IC*KS], q = kpos*IC + ic
__global__ void k_convw_t16(const float* __restrict__ w, _Float16* __restrict__ d,
                            int OC, int IC) {
  int K = IC * KS;
  long i = (long)blockIdx.x * blockDim.x + threadIdx.x;
  if (i >= (long)OC * K) return;
  int oc = (int)(i / K), q = (int)(i - (long)oc * K);
  int kpos = q / IC, ic = q - kpos * IC;
  d[i] = (_Float16)w[((long)oc * IC + ic) * KS + kpos];
}
// embedding gather + cast: xt0[b][l][e] = emb[tok][e]
__global__ void k_embed(const int* __restrict__ tok, const float* __restrict__ emb,
                        _Float16* __restrict__ d) {
  long i = (long)blockIdx.x * blockDim.x + threadIdx.x;  // over B*L
  if (i >= (long)BB * LL) return;
  int tv = tok[i];
  const float* e = emb + (long)tv * EMB;
  _Float16* o = d + i * EMB;
  for (int c = 0; c < EMB; ++c) o[c] = (_Float16)e[c];
}
// per-node per-head attention scores
__global__ void k_att_scores(const _Float16* __restrict__ h, int ldh,
                             const float* __restrict__ ats, const float* __restrict__ atd,
                             float* __restrict__ as_, float* __restrict__ ad_,
                             int H, int Cc) {
  long i = (long)blockIdx.x * blockDim.x + threadIdx.x;  // over NN*H
  if (i >= (long)NN * H) return;
  int n = (int)(i / H), hh = (int)(i - (long)n * H);
  const _Float16* row = h + (long)n * ldh + hh * Cc;
  float ss = 0.f, sd = 0.f;
  for (int c = 0; c < Cc; ++c) {
    float hv = (float)row[c];
    ss += hv * ats[hh * Cc + c];
    sd += hv * atd[hh * Cc + c];
  }
  as_[i] = ss; ad_[i] = sd;
}
__device__ __forceinline__ void atomicMaxF(float* addr, float v) {
  if (v >= 0.f) atomicMax((int*)addr, __float_as_int(v));
  else          atomicMin((unsigned*)addr, __float_as_uint(v));
}
__global__ void k_edge_alpha(const int* __restrict__ src, const int* __restrict__ dst,
                             const float* __restrict__ as_, const float* __restrict__ ad_,
                             float* __restrict__ alpha, float* __restrict__ mx, int H) {
  long i = (long)blockIdx.x * blockDim.x + threadIdx.x;  // over ETOT*H
  if (i >= (long)ETOT * H) return;
  int e = (int)(i / H), hh = (int)(i - (long)e * H);
  int s, d;
  if (e < EE) { s = src[e]; d = dst[e]; } else { s = d = e - EE; }
  float al = as_[(long)s * H + hh] + ad_[(long)d * H + hh];
  al = al > 0.f ? al : 0.2f * al;          // leaky_relu 0.2
  alpha[i] = al;
  atomicMaxF(&mx[(long)d * H + hh], al);
}
__global__ void k_edge_exp(const int* __restrict__ dst, float* __restrict__ alpha,
                           const float* __restrict__ mx, float* __restrict__ z, int H) {
  long i = (long)blockIdx.x * blockDim.x + threadIdx.x;
  if (i >= (long)ETOT * H) return;
  int e = (int)(i / H), hh = (int)(i - (long)e * H);
  int d = (e < EE) ? dst[e] : (e - EE);
  float a = expf(alpha[i] - mx[(long)d * H + hh]);
  alpha[i] = a;
  atomicAdd(&z[(long)d * H + hh], a);
}
__global__ void k_edge_scatter(const int* __restrict__ src, const int* __restrict__ dst,
                               const float* __restrict__ alpha, const float* __restrict__ z,
                               const _Float16* __restrict__ h, int ldh,
                               float* __restrict__ out, int H, int Cc, int ostride) {
  long i = (long)blockIdx.x * blockDim.x + threadIdx.x;  // over ETOT*H
  if (i >= (long)ETOT * H) return;
  int e = (int)(i / H), hh = (int)(i - (long)e * H);
  int s, d;
  if (e < EE) { s = src[e]; d = dst[e]; } else { s = d = e - EE; }
  float w = alpha[i] / z[(long)d * H + hh];
  const _Float16* hs = h + (long)s * ldh + hh * Cc;
  float* o = out + (long)d * ostride + hh * Cc;
  for (int c = 0; c < Cc; ++c) atomicAdd(&o[c], w * (float)hs[c]);
}
// elu(out1 + b1) -> f16 padded to 832 cols
__global__ void k_finalize_gat1(const float* __restrict__ o1, const float* __restrict__ b1,
                                _Float16* __restrict__ d) {
  long i = (long)blockIdx.x * blockDim.x + threadIdx.x;  // over NN*780
  if (i >= (long)NN * (HH * FD)) return;
  int n = (int)(i / (HH * FD)), j = (int)(i - (long)n * (HH * FD));
  float v = o1[i] + b1[j];
  v = v > 0.f ? v : (expf(v) - 1.f);
  d[(long)n * 832 + j] = (_Float16)v;
}
// relu(out2 + b2) then max over 200 nodes per graph -> f16 [B][OD]
__global__ void k_pool_graph(const float* __restrict__ o2, const float* __restrict__ b2,
                             _Float16* __restrict__ d) {
  long i = (long)blockIdx.x * blockDim.x + threadIdx.x;  // over B*OD
  if (i >= (long)BB * OD) return;
  int b = (int)(i / OD), od = (int)(i - (long)b * OD);
  float bias = b2[od];
  float mx = 0.f;                                  // relu floor
  for (int n = 0; n < NPG; ++n) {
    float v = o2[((long)b * NPG + n) * OD + od] + bias;
    v = v > 0.f ? v : 0.f;
    mx = fmaxf(mx, v);
  }
  d[i] = (_Float16)mx;
}
// global max pool over protein length (values already >=0) -> f16 [B][ld]
__global__ void k_pool_prot(const _Float16* __restrict__ x, _Float16* __restrict__ d,
                            int Lout, int OC, int ld) {
  long i = (long)blockIdx.x * blockDim.x + threadIdx.x;  // over B*OC
  if (i >= (long)BB * OC) return;
  int b = (int)(i / OC), oc = (int)(i - (long)b * OC);
  float mx = 0.f;
  for (int t = 0; t < Lout; ++t)
    mx = fmaxf(mx, (float)x[((long)b * Lout + t) * OC + oc]);
  d[(long)b * ld + oc] = (_Float16)mx;
}
// final head: d_out[b] = fc2o[b,:] . out_w + out_b
__global__ void k_final(const _Float16* __restrict__ x, const float* __restrict__ w,
                        const float* __restrict__ b, float* __restrict__ out) {
  int bb = threadIdx.x;
  if (bb >= BB) return;
  float s = 0.f;
  const _Float16* row = x + (long)bb * 256;
  for (int k = 0; k < 256; ++k) s += (float)row[k] * w[k];
  out[bb] = s + b[0];
}

// ---------------------------------------------------------------- host
static inline unsigned gb(long n) { return (unsigned)((n + 255) / 256); }

extern "C" void kernel_launch(void* const* d_in, const int* in_sizes, int n_in,
                              void* d_out, int out_size, void* d_ws, size_t ws_size,
                              hipStream_t stream) {
  const float* x     = (const float*)d_in[0];
  const float* W1    = (const float*)d_in[1];
  const float* as1w  = (const float*)d_in[2];
  const float* ad1w  = (const float*)d_in[3];
  const float* b1    = (const float*)d_in[4];
  const float* W2    = (const float*)d_in[5];
  const float* as2w  = (const float*)d_in[6];
  const float* ad2w  = (const float*)d_in[7];
  const float* b2    = (const float*)d_in[8];
  const float* fcg1w = (const float*)d_in[9];
  const float* fcg1b = (const float*)d_in[10];
  const float* emb   = (const float*)d_in[11];
  const float* cw0   = (const float*)d_in[12];
  const float* cb0   = (const float*)d_in[13];
  const float* bng0  = (const float*)d_in[14];
  const float* bnb0  = (const float*)d_in[15];
  const float* cw1   = (const float*)d_in[16];
  const float* cb1   = (const float*)d_in[17];
  const float* bng1  = (const float*)d_in[18];
  const float* bnb1  = (const float*)d_in[19];
  const float* cw2   = (const float*)d_in[20];
  const float* cb2   = (const float*)d_in[21];
  const float* bng2  = (const float*)d_in[22];
  const float* bnb2  = (const float*)d_in[23];
  const float* fcxtw = (const float*)d_in[24];
  const float* fcxtb = (const float*)d_in[25];
  const float* bnfg  = (const float*)d_in[26];
  const float* bnfb  = (const float*)d_in[27];
  const float* fc1w  = (const float*)d_in[28];
  const float* fc1b  = (const float*)d_in[29];
  const float* fc2w  = (const float*)d_in[30];
  const float* fc2b  = (const float*)d_in[31];
  const float* outw  = (const float*)d_in[32];
  const float* outb  = (const float*)d_in[33];
  const int*   eidx  = (const int*)d_in[34];
  const int*   tenc  = (const int*)d_in[36];
  const int* esrc = eidx;
  const int* edst = eidx + EE;

  // ---- workspace carve (256B aligned) ----
  char* base = (char*)d_ws;
  size_t off = 0;
  auto alloc = [&](size_t bytes) -> void* {
    void* p = base + off;
    off += (bytes + 255) & ~(size_t)255;
    return p;
  };
  _Float16* x16    = (_Float16*)alloc((size_t)NN * 128 * 2);
  _Float16* W1t    = (_Float16*)alloc((size_t)780 * 128 * 2);
  _Float16* h1     = (_Float16*)alloc((size_t)NN * 832 * 2);
  float*    as1    = (float*)alloc((size_t)NN * HH * 4);
  float*    ad1    = (float*)alloc((size_t)NN * HH * 4);
  float*    m1     = (float*)alloc((size_t)NN * HH * 4);
  float*    z1     = (float*)alloc((size_t)NN * HH * 4);
  float*    alpha1 = (float*)alloc((size_t)ETOT * HH * 4);
  float*    out1   = (float*)alloc((size_t)NN * 780 * 4);
  _Float16* g2in   = (_Float16*)alloc((size_t)NN * 832 * 2);
  _Float16* W2t    = (_Float16*)alloc((size_t)128 * 832 * 2);
  _Float16* h2     = (_Float16*)alloc((size_t)NN * 128 * 2);
  float*    as2    = (float*)alloc((size_t)NN * 4);
  float*    ad2    = (float*)alloc((size_t)NN * 4);
  float*    m2     = (float*)alloc((size_t)NN * 4);
  float*    z2     = (float*)alloc((size_t)NN * 4);
  float*    alpha2 = (float*)alloc((size_t)ETOT * 4);
  float*    out2   = (float*)alloc((size_t)NN * 128 * 4);
  _Float16* xg16   = (_Float16*)alloc((size_t)BB * 128 * 2);
  _Float16* fcg1t  = (_Float16*)alloc((size_t)128 * 128 * 2);
  _Float16* cat16  = (_Float16*)alloc((size_t)BB * 256 * 2);
  _Float16* xt0    = (_Float16*)alloc((size_t)BB * LL * EMB * 2);
  _Float16* c0t    = (_Float16*)alloc((size_t)32 * 2048 * 2);
  _Float16* xt1    = (_Float16*)alloc((size_t)BB * 985 * 32 * 2);
  _Float16* c1t    = (_Float16*)alloc((size_t)64 * 512 * 2);
  _Float16* xt2    = (_Float16*)alloc((size_t)BB * 970 * 64 * 2);
  _Float16* c2t    = (_Float16*)alloc((size_t)96 * 1024 * 2);
  _Float16* xt3    = (_Float16*)alloc((size_t)BB * 955 * 96 * 2);
  _Float16* xtp16  = (_Float16*)alloc((size_t)BB * 128 * 2);
  _Float16* fcxtt  = (_Float16*)alloc((size_t)128 * 128 * 2);
  _Float16* fc1t   = (_Float16*)alloc((size_t)1024 * 256 * 2);
  _Float16* fc1o   = (_Float16*)alloc((size_t)BB * 1024 * 2);
  _Float16* fc2t   = (_Float16*)alloc((size_t)256 * 1024 * 2);
  _Float16* fc2o   = (_Float16*)alloc((size_t)BB * 256 * 2);
  (void)ws_size; (void)in_sizes; (void)n_in; (void)out_size;

  const int T = 256;

  // ---- weight prep (f16, transposed, K padded to mult of 64) ----
  k_cast_pad16   <<<gb((long)NN * 128), T, 0, stream>>>(x, x16, NN, FD, 128);
  k_w_transpose16<<<gb((long)780 * 128), T, 0, stream>>>(W1, W1t, 78, 780, 128);
  k_w_transpose16<<<gb((long)128 * 832), T, 0, stream>>>(W2, W2t, 780, 128, 832);
  k_w_transpose16<<<gb((long)128 * 128), T, 0, stream>>>(fcg1w, fcg1t, 128, 128, 128);
  k_w_transpose16<<<gb((long)128 * 128), T, 0, stream>>>(fcxtw, fcxtt, 96, 128, 128);
  k_w_transpose16<<<gb((long)1024 * 256), T, 0, stream>>>(fc1w, fc1t, 256, 1024, 256);
  k_w_transpose16<<<gb((long)256 * 1024), T, 0, stream>>>(fc2w, fc2t, 1024, 256, 1024);
  k_convw_t16    <<<gb((long)32 * 2048), T, 0, stream>>>(cw0, c0t, 32, 128);
  k_convw_t16    <<<gb((long)64 * 512),  T, 0, stream>>>(cw1, c1t, 64, 32);
  k_convw_t16    <<<gb((long)96 * 1024), T, 0, stream>>>(cw2, c2t, 96, 64);

  // ---- zero / init fills (every launch: deterministic) ----
  k_fill_u32<<<gb((long)NN * 416), T, 0, stream>>>((unsigned*)h1,  (long)NN * 416, 0u);
  k_fill_u32<<<gb((long)NN * 416), T, 0, stream>>>((unsigned*)g2in,(long)NN * 416, 0u);
  k_fill_u32<<<gb((long)BB * 64), T, 0, stream>>>((unsigned*)xtp16, (long)BB * 64, 0u);
  k_fill_f32<<<gb((long)NN * 780), T, 0, stream>>>(out1, (long)NN * 780, 0.f);
  k_fill_f32<<<gb((long)NN * 128), T, 0, stream>>>(out2, (long)NN * 128, 0.f);
  k_fill_f32<<<gb((long)NN * HH), T, 0, stream>>>(m1, (long)NN * HH, -1e30f);
  k_fill_f32<<<gb((long)NN * HH), T, 0, stream>>>(z1, (long)NN * HH, 0.f);
  k_fill_f32<<<gb((long)NN), T, 0, stream>>>(m2, (long)NN, -1e30f);
  k_fill_f32<<<gb((long)NN), T, 0, stream>>>(z2, (long)NN, 0.f);

  // ---- GAT layer 1: h1 = x @ W1 ----
  {
    dim3 g((780 + BN - 1) / BN, (NN + BM - 1) / BM);
    gemm_wmma<0, 0><<<g, T, 0, stream>>>(x16, W1t, nullptr, nullptr, nullptr,
                                         h1, NN, 780, 128, 832, 0, 0, 0);
  }
  k_att_scores <<<gb((long)NN * HH), T, 0, stream>>>(h1, 832, as1w, ad1w, as1, ad1, HH, FD);
  k_edge_alpha <<<gb((long)ETOT * HH), T, 0, stream>>>(esrc, edst, as1, ad1, alpha1, m1, HH);
  k_edge_exp   <<<gb((long)ETOT * HH), T, 0, stream>>>(edst, alpha1, m1, z1, HH);
  k_edge_scatter<<<gb((long)ETOT * HH), T, 0, stream>>>(esrc, edst, alpha1, z1, h1, 832,
                                                        out1, HH, FD, 780);
  k_finalize_gat1<<<gb((long)NN * 780), T, 0, stream>>>(out1, b1, g2in);

  // ---- GAT layer 2: h2 = elu(out1) @ W2 ----
  {
    dim3 g((128 + BN - 1) / BN, (NN + BM - 1) / BM);
    gemm_wmma<0, 0><<<g, T, 0, stream>>>(g2in, W2t, nullptr, nullptr, nullptr,
                                         h2, NN, 128, 832, 128, 0, 0, 0);
  }
  k_att_scores <<<gb((long)NN), T, 0, stream>>>(h2, 128, as2w, ad2w, as2, ad2, 1, 128);
  k_edge_alpha <<<gb((long)ETOT), T, 0, stream>>>(esrc, edst, as2, ad2, alpha2, m2, 1);
  k_edge_exp   <<<gb((long)ETOT), T, 0, stream>>>(edst, alpha2, m2, z2, 1);
  k_edge_scatter<<<gb((long)ETOT), T, 0, stream>>>(esrc, edst, alpha2, z2, h2, 128,
                                                   out2, 1, 128, 128);
  k_pool_graph <<<gb((long)BB * OD), T, 0, stream>>>(out2, b2, xg16);

  // ---- fc_g1: relu(xg @ W + b) -> cat16[:, :128] ----
  {
    dim3 g((128 + BN - 1) / BN, (BB + BM - 1) / BM);
    gemm_wmma<0, 1><<<g, T, 0, stream>>>(xg16, fcg1t, fcg1b, nullptr, nullptr,
                                         cat16, BB, 128, 128, 256, 0, 0, 0);
  }

  // ---- protein branch ----
  k_embed<<<gb((long)BB * LL), T, 0, stream>>>(tenc, emb, xt0);
  {  // conv1: IC=128, K=2048, Lout=985, OC=32   (implicit GEMM, bn+relu fused)
    dim3 g((32 + BN - 1) / BN, (BB * 985 + BM - 1) / BM);
    gemm_wmma<1, 3><<<g, T, 0, stream>>>(xt0, c0t, cb0, bng0, bnb0,
                                         xt1, BB * 985, 32, 2048, 32, 128, 1000, 985);
  }
  {  // conv2: IC=32, K=512, Lout=970, OC=64
    dim3 g((64 + BN - 1) / BN, (BB * 970 + BM - 1) / BM);
    gemm_wmma<1, 3><<<g, T, 0, stream>>>(xt1, c1t, cb1, bng1, bnb1,
                                         xt2, BB * 970, 64, 512, 64, 32, 985, 970);
  }
  {  // conv3: IC=64, K=1024, Lout=955, OC=96
    dim3 g((96 + BN - 1) / BN, (BB * 955 + BM - 1) / BM);
    gemm_wmma<1, 3><<<g, T, 0, stream>>>(xt2, c2t, cb2, bng2, bnb2,
                                         xt3, BB * 955, 96, 1024, 96, 64, 970, 955);
  }
  k_pool_prot<<<gb((long)BB * 96), T, 0, stream>>>(xt3, xtp16, 955, 96, 128);
  {  // fc_xt + bn + relu -> cat16[:, 128:]  (K padded 96->128, pads zero)
    dim3 g((128 + BN - 1) / BN, (BB + BM - 1) / BM);
    gemm_wmma<0, 3><<<g, T, 0, stream>>>(xtp16, fcxtt, fcxtb, bnfg, bnfb,
                                         cat16 + 128, BB, 128, 128, 256, 0, 0, 0);
  }

  // ---- fusion head ----
  {
    dim3 g((1024 + BN - 1) / BN, (BB + BM - 1) / BM);
    gemm_wmma<0, 1><<<g, T, 0, stream>>>(cat16, fc1t, fc1b, nullptr, nullptr,
                                         fc1o, BB, 1024, 256, 1024, 0, 0, 0);
  }
  {
    dim3 g((256 + BN - 1) / BN, (BB + BM - 1) / BM);
    gemm_wmma<0, 1><<<g, T, 0, stream>>>(fc1o, fc2t, fc2b, nullptr, nullptr,
                                         fc2o, BB, 256, 1024, 256, 0, 0, 0);
  }
  k_final<<<1, 128, 0, stream>>>(fc2o, outw, outb, (float*)d_out);
}
